// LinearAttention_2705829396867
// MI455X (gfx1250) — matrix operand order repaired and checked by reading, not compiled, not run on hardware
//
#include <hip/hip_runtime.h>

// ---------------------------------------------------------------------------
// LinearAttention fused pipeline for gfx1250 (MI455X)
// bf16 WMMA + f32 accumulate, async global->LDS staging, layout-native frags
// ---------------------------------------------------------------------------

#define BATCH    8
#define DIM      512
#define SEQ      4096
#define HEADS    8
#define DHEAD    64
#define TRIPLE   1536     // 3 * HIDDEN
#define SCALE    0.125f   // 64^-0.5
#define EPS      1e-5f

typedef __attribute__((ext_vector_type(16))) __bf16 v16bf;
typedef __attribute__((ext_vector_type(8)))  __bf16 v8bf;
typedef __attribute__((ext_vector_type(4)))  __bf16 v4bf;
typedef __attribute__((ext_vector_type(8)))  float  v8f;

__device__ __forceinline__ v8f zero8() {
    v8f z;
    #pragma unroll
    for (int i = 0; i < 8; ++i) z[i] = 0.0f;
    return z;
}

// CDNA5 async copy: 16 bytes global -> LDS, tracked by ASYNCcnt (no VGPR data).
__device__ __forceinline__ void async_copy_b128(unsigned int lds_off, const __bf16* src) {
    asm volatile("global_load_async_to_lds_b128 %0, %1, off"
                 :: "v"(lds_off), "v"(src) : "memory");
}
__device__ __forceinline__ void wait_async() {
    asm volatile("s_wait_asynccnt 0x0" ::: "memory");
}
__device__ __forceinline__ unsigned int lds_off32(const void* p) {
    return (unsigned int)(unsigned long long)p;   // low 32 bits = LDS offset
}

// A fragment (16x32 bf16, M x K), source row-major [M][K], 'stride' elems/row.
// lanes 0-15 -> M=lane, K {0..7,16..23}; lanes 16-31 -> M=lane-16, K {8..15,24..31}.
__device__ __forceinline__ v16bf load_a_frag(const __bf16* base, int stride, int lane) {
    int m  = lane & 15;
    int ko = (lane < 16) ? 0 : 8;
    const __bf16* p = base + (size_t)m * stride + ko;
    v8bf lo = *(const v8bf*)(p);
    v8bf hi = *(const v8bf*)(p + 16);
    return __builtin_shufflevector(lo, hi, 0,1,2,3,4,5,6,7,8,9,10,11,12,13,14,15);
}

// B fragment (32x16 bf16, K x N), source [N][K] (K contiguous), 'stride' elems
// between N columns. lanes 0-15: N=lane, K {0..7,16..23}; lanes 16-31: N=lane-16, K {8..15,24..31}.
__device__ __forceinline__ v16bf load_b_frag(const __bf16* base, int stride, int lane) {
    int n  = lane & 15;
    int ko = (lane < 16) ? 0 : 8;
    const __bf16* p = base + (size_t)n * stride + ko;
    v8bf lo = *(const v8bf*)(p);
    v8bf hi = *(const v8bf*)(p + 16);
    return __builtin_shufflevector(lo, hi, 0,1,2,3,4,5,6,7,8,9,10,11,12,13,14,15);
}

__device__ __forceinline__ v8f wmma_bf16(v16bf a, v16bf b, v8f c) {
    return __builtin_amdgcn_wmma_f32_16x16x32_bf16(false, a, false, b, (short)0, c, false, false);
}

// ---------------------------------------------------------------------------
// P0a: x[b][c][n] f32  ->  xT[b][n][c] bf16   (tiled transpose + convert, once)
// ---------------------------------------------------------------------------
__global__ void k_xpose(const float* __restrict__ x, __bf16* __restrict__ xT) {
    __shared__ float tile[32][33];
    const int tid = threadIdx.x;
    const int b = blockIdx.z, c0 = blockIdx.y * 32, n0 = blockIdx.x * 32;

    int c = tid >> 3, ns = (tid & 7) << 2;
    float4 f = *(const float4*)(x + ((size_t)b * DIM + c0 + c) * SEQ + n0 + ns);
    tile[c][ns + 0] = f.x; tile[c][ns + 1] = f.y;
    tile[c][ns + 2] = f.z; tile[c][ns + 3] = f.w;
    __syncthreads();

    int n = tid >> 3, cs = (tid & 7) << 2;
    v4bf o;
    #pragma unroll
    for (int j = 0; j < 4; ++j) o[j] = (__bf16)tile[cs + j][n];
    *(v4bf*)(xT + ((size_t)b * SEQ + n0 + n) * DIM + c0 + cs) = o;
}

// P0b: w_qkv f32 -> bf16 (1536x512, row-major kept)
__global__ void k_wconv(const float* __restrict__ w, __bf16* __restrict__ wb) {
    int id = (blockIdx.x * 256 + threadIdx.x) * 4;
    float4 f = *(const float4*)(w + id);
    v4bf o; o[0] = (__bf16)f.x; o[1] = (__bf16)f.y; o[2] = (__bf16)f.z; o[3] = (__bf16)f.w;
    *(v4bf*)(wb + id) = o;
}

// ---------------------------------------------------------------------------
// Kernel 1: qkv = W_qkv @ x per batch. Tile M=128, N=128, K-step 32.
// 8 waves, each wave: M=16 x N=128 (8 WMMA tiles). A direct from global bf16,
// B tile async-copied global->LDS (already transposed layout, zero VALU).
// Epilogues: q -> softmax(d) * SCALE -> qT (transposed, packed b128 stores)
//            k -> exp -> kvb rows 0..511, atomic row sums -> ksum
//            v -> kvb rows 512..1023
// ---------------------------------------------------------------------------
__global__ void k_qkv_gemm(const __bf16* __restrict__ xT,
                           const __bf16* __restrict__ wb,
                           __bf16* __restrict__ qT,
                           __bf16* __restrict__ kvb,
                           float* __restrict__ ksum) {
    __shared__ __bf16 ldsB[128][40];   // [n][k]
    __shared__ float  colsum[2][128];

    const int tid  = threadIdx.x;
    const int lane = tid & 31;
    const int mb   = tid >> 5;            // m-block 0..7 (16 rows each)
    const int b  = blockIdx.z;
    const int m0 = blockIdx.y * 128;
    const int n0 = blockIdx.x * 128;

    v8f acc[8];
    #pragma unroll
    for (int i = 0; i < 8; ++i) acc[i] = zero8();

    for (int kc = 0; kc < DIM; kc += 32) {
        // async stage: xT tile [128 n][32 c] -> LDS, 2 x 16B per thread
        #pragma unroll
        for (int i = 0; i < 2; ++i) {
            int seg = tid + i * 256;
            int n   = seg >> 2;
            int cs  = (seg & 3) << 3;
            async_copy_b128(lds_off32(&ldsB[n][cs]),
                            xT + ((size_t)b * SEQ + n0 + n) * DIM + kc + cs);
        }
        wait_async();
        __syncthreads();

        v16bf afrag = load_a_frag(wb + (size_t)(m0 + mb * 16) * DIM + kc, DIM, lane);
        #pragma unroll
        for (int nt = 0; nt < 8; ++nt) {
            v16bf bfrag = load_b_frag(&ldsB[nt * 16][0], 40, lane);
            acc[nt] = wmma_bf16(afrag, bfrag, acc[nt]);
        }
        __syncthreads();
    }

    const int half   = (lane >= 16) ? 8 : 0;
    const int region = m0 >> 9;   // 0=q, 1=k, 2=v

    if (region == 0) {
        #pragma unroll
        for (int nt = 0; nt < 8; ++nt)
            #pragma unroll
            for (int r = 0; r < 8; ++r) acc[nt][r] = __expf(acc[nt][r]);

        colsum[tid >> 7][tid & 127] = 0.0f;
        __syncthreads();
        const int grp = mb >> 2;   // two 64-row head groups in this tile
        #pragma unroll
        for (int nt = 0; nt < 8; ++nt) {
            float s = 0.0f;
            #pragma unroll
            for (int r = 0; r < 8; ++r) s += acc[nt][r];
            s += __shfl_xor(s, 16, 32);
            if (lane < 16) atomicAdd(&colsum[grp][nt * 16 + lane], s);
        }
        __syncthreads();
        #pragma unroll
        for (int nt = 0; nt < 8; ++nt) {
            int col   = nt * 16 + (lane & 15);
            float inv = SCALE / colsum[grp][col];
            v8bf pack;
            #pragma unroll
            for (int r = 0; r < 8; ++r) pack[r] = (__bf16)(acc[nt][r] * inv);
            int cbase = m0 + mb * 16 + half;            // contiguous 8 channels
            *(v8bf*)(qT + ((size_t)b * SEQ + n0 + col) * DIM + cbase) = pack;
        }
    } else if (region == 1) {
        #pragma unroll
        for (int nt = 0; nt < 8; ++nt)
            #pragma unroll
            for (int r = 0; r < 8; ++r) acc[nt][r] = __expf(acc[nt][r]);

        float s[8];
        #pragma unroll
        for (int r = 0; r < 8; ++r) {
            s[r] = 0.0f;
            #pragma unroll
            for (int nt = 0; nt < 8; ++nt) s[r] += acc[nt][r];
            s[r] += __shfl_xor(s[r], 1, 32);
            s[r] += __shfl_xor(s[r], 2, 32);
            s[r] += __shfl_xor(s[r], 4, 32);
            s[r] += __shfl_xor(s[r], 8, 32);
        }
        if ((lane & 15) == 0) {
            int rbase = (m0 - 512) + mb * 16 + half;
            #pragma unroll
            for (int r = 0; r < 8; ++r)
                atomicAdd(&ksum[b * DIM + rbase + r], s[r]);
        }
        #pragma unroll
        for (int nt = 0; nt < 8; ++nt)
            #pragma unroll
            for (int r = 0; r < 8; ++r) {
                int row = (m0 - 512) + mb * 16 + r + half;
                kvb[((size_t)b * 1024 + row) * SEQ + n0 + nt * 16 + (lane & 15)] =
                    (__bf16)acc[nt][r];
            }
    } else {
        #pragma unroll
        for (int nt = 0; nt < 8; ++nt)
            #pragma unroll
            for (int r = 0; r < 8; ++r) {
                int row = (m0 - 512) + mb * 16 + r + half;   // v rows 512..1023
                kvb[((size_t)b * 1024 + row) * SEQ + n0 + nt * 16 + (lane & 15)] =
                    (__bf16)acc[nt][r];
            }
    }
}

// ---------------------------------------------------------------------------
// Kernel 2: ctx_raw[b,h,d,e] += sum_n exp_k[d,n] * v[e,n]   (K split 4-way)
// One workgroup per (b,h,slice); 4 waves, each M=64 x N=16, K=1024.
// ---------------------------------------------------------------------------
__global__ void k_context(const __bf16* __restrict__ kvb, float* __restrict__ ctx) {
    const int tid  = threadIdx.x;
    const int lane = tid & 31;
    const int wid  = tid >> 5;         // e-tile 0..3
    const int h = blockIdx.x, b = blockIdx.y, sl = blockIdx.z;

    const __bf16* kb = kvb + ((size_t)b * 1024 + h * DHEAD) * SEQ;
    const __bf16* vb = kvb + ((size_t)b * 1024 + 512 + h * DHEAD) * SEQ;

    v8f acc[4] = { zero8(), zero8(), zero8(), zero8() };

    const int kend = (sl + 1) * 1024;
    for (int kc = sl * 1024; kc < kend; kc += 32) {
        v16bf bfrag = load_b_frag(vb + (size_t)(wid * 16) * SEQ + kc, SEQ, lane);
        #pragma unroll
        for (int mt = 0; mt < 4; ++mt) {
            v16bf afrag = load_a_frag(kb + (size_t)(mt * 16) * SEQ + kc, SEQ, lane);
            acc[mt] = wmma_bf16(afrag, bfrag, acc[mt]);
        }
    }

    const int half = (lane >= 16) ? 8 : 0;
    const int e    = wid * 16 + (lane & 15);
    #pragma unroll
    for (int mt = 0; mt < 4; ++mt)
        #pragma unroll
        for (int r = 0; r < 8; ++r) {
            int d = mt * 16 + r + half;
            atomicAdd(&ctx[(((size_t)(b * HEADS + h) * DHEAD + d) * DHEAD) + e],
                      acc[mt][r]);
        }
}

// ---------------------------------------------------------------------------
// Kernel 3 (tiny): Wf[b][o][h*64+d] = (1/ksum[d]) * sum_e W_out[o][h*64+e]*ctx
// ---------------------------------------------------------------------------
__global__ void k_wfuse(const float* __restrict__ wout,
                        const float* __restrict__ ctx,
                        const float* __restrict__ ksum,
                        __bf16* __restrict__ wfused) {
    int id  = blockIdx.x * 256 + threadIdx.x;
    int col = id & 511;
    int o   = (id >> 9) & 511;
    int b   = id >> 18;
    int h   = col >> 6;
    int d   = col & 63;
    const float* wrow = wout + (size_t)o * DIM + h * DHEAD;
    const float* crow = ctx + ((size_t)(b * HEADS + h) * DHEAD + d) * DHEAD;
    float acc = 0.0f;
    #pragma unroll 8
    for (int e = 0; e < DHEAD; ++e) acc += wrow[e] * crow[e];
    float inv = 1.0f / ksum[b * DIM + col];
    wfused[((size_t)b * DIM + o) * DIM + col] = (__bf16)(acc * inv);
}

// ---------------------------------------------------------------------------
// Kernel 4: y = Wf[b] @ qT[b]^T + bias, channel LayerNorm * g.
// 8 waves x 64 rows = full M=512 in one workgroup; N=64. B tile async-staged.
// ---------------------------------------------------------------------------
__global__ void k_final(const __bf16* __restrict__ qT,
                        const __bf16* __restrict__ wfused,
                        const float* __restrict__ bias,
                        const float* __restrict__ g,
                        float* __restrict__ out) {
    __shared__ __bf16 ldsQ[64][40];
    __shared__ float  psum[8][64];
    __shared__ float  psq[8][64];
    __shared__ float  cmean[64];
    __shared__ float  crstd[64];

    const int tid  = threadIdx.x;
    const int lane = tid & 31;
    const int wid  = tid >> 5;     // 64-row block
    const int b    = blockIdx.y;
    const int n0   = blockIdx.x * 64;

    v8f acc[4][4];
    #pragma unroll
    for (int mt = 0; mt < 4; ++mt)
        #pragma unroll
        for (int nt = 0; nt < 4; ++nt) acc[mt][nt] = zero8();

    const __bf16* wf = wfused + (size_t)b * DIM * DIM;

    for (int kc = 0; kc < DIM; kc += 32) {
        {   // async stage: qT tile [64 n][32 c] -> LDS, one 16B per thread
            int n  = tid >> 2;
            int cs = (tid & 3) << 3;
            async_copy_b128(lds_off32(&ldsQ[n][cs]),
                            qT + ((size_t)b * SEQ + n0 + n) * DIM + kc + cs);
        }
        wait_async();
        __syncthreads();

        v16bf afrag[4], bfrag[4];
        #pragma unroll
        for (int mt = 0; mt < 4; ++mt)
            afrag[mt] = load_a_frag(wf + (size_t)(wid * 64 + mt * 16) * DIM + kc, DIM, lane);
        #pragma unroll
        for (int nt = 0; nt < 4; ++nt)
            bfrag[nt] = load_b_frag(&ldsQ[nt * 16][0], 40, lane);
        #pragma unroll
        for (int mt = 0; mt < 4; ++mt)
            #pragma unroll
            for (int nt = 0; nt < 4; ++nt)
                acc[mt][nt] = wmma_bf16(afrag[mt], bfrag[nt], acc[mt][nt]);
        __syncthreads();
    }

    const int half = (lane >= 16) ? 8 : 0;

    #pragma unroll
    for (int nt = 0; nt < 4; ++nt) {
        float s = 0.0f, sq = 0.0f;
        #pragma unroll
        for (int mt = 0; mt < 4; ++mt)
            #pragma unroll
            for (int r = 0; r < 8; ++r) {
                int row = wid * 64 + mt * 16 + r + half;
                float y = acc[mt][nt][r] + bias[row];
                acc[mt][nt][r] = y;
                s  += y;
                sq += y * y;
            }
        s  += __shfl_xor(s, 16, 32);
        sq += __shfl_xor(sq, 16, 32);
        if (lane < 16) { psum[wid][nt * 16 + lane] = s; psq[wid][nt * 16 + lane] = sq; }
    }
    __syncthreads();

    if (tid < 64) {
        float S = 0.0f, Q = 0.0f;
        #pragma unroll
        for (int w = 0; w < 8; ++w) { S += psum[w][tid]; Q += psq[w][tid]; }
        float mean = S * (1.0f / DIM);
        float var  = Q * (1.0f / DIM) - mean * mean;
        cmean[tid] = mean;
        crstd[tid] = rsqrtf(var + EPS);
    }
    __syncthreads();

    #pragma unroll
    for (int mt = 0; mt < 4; ++mt)
        #pragma unroll
        for (int nt = 0; nt < 4; ++nt)
            #pragma unroll
            for (int r = 0; r < 8; ++r) {
                int row = wid * 64 + mt * 16 + r + half;
                int col = nt * 16 + (lane & 15);
                float y = (acc[mt][nt][r] - cmean[col]) * crstd[col] * g[row];
                out[((size_t)b * DIM + row) * SEQ + n0 + col] = y;
            }
}

// ---------------------------------------------------------------------------
// Launch
// ---------------------------------------------------------------------------
extern "C" void kernel_launch(void* const* d_in, const int* in_sizes, int n_in,
                              void* d_out, int out_size, void* d_ws, size_t ws_size,
                              hipStream_t stream) {
    const float* x     = (const float*)d_in[0];
    const float* wqkv  = (const float*)d_in[1];
    const float* wout  = (const float*)d_in[2];
    const float* bout  = (const float*)d_in[3];
    const float* g     = (const float*)d_in[4];
    float* out = (float*)d_out;

    char* ws = (char*)d_ws;
    size_t off = 0;
    const size_t xT_bytes   = (size_t)BATCH * SEQ * DIM * sizeof(__bf16);      // 32 MB
    const size_t qT_bytes   = (size_t)BATCH * SEQ * DIM * sizeof(__bf16);      // 32 MB
    const size_t kvb_bytes  = (size_t)BATCH * 1024 * SEQ * sizeof(__bf16);     // 64 MB
    const size_t wb_bytes   = (size_t)TRIPLE * DIM * sizeof(__bf16);           // 1.5 MB
    const size_t ksum_bytes = (size_t)BATCH * DIM * sizeof(float);             // 16 KB
    const size_t ctx_bytes  = (size_t)BATCH * HEADS * DHEAD * DHEAD * sizeof(float);

    __bf16* xT     = (__bf16*)(ws + off); off += xT_bytes;
    __bf16* qT     = (__bf16*)(ws + off); off += qT_bytes;
    __bf16* kvb    = (__bf16*)(ws + off); off += kvb_bytes;
    __bf16* wb     = (__bf16*)(ws + off); off += wb_bytes;
    float*  ksum   = (float*)(ws + off);  off += ksum_bytes;
    float*  ctx    = (float*)(ws + off);  off += ctx_bytes;
    __bf16* wfused = (__bf16*)(ws + off);

    hipMemsetAsync(ksum, 0, ksum_bytes + ctx_bytes, stream);   // ksum + ctx contiguous

    k_wconv<<<(TRIPLE * DIM) / 1024, 256, 0, stream>>>(wqkv, wb);
    k_xpose<<<dim3(SEQ / 32, DIM / 32, BATCH), 256, 0, stream>>>(x, xT);
    k_qkv_gemm<<<dim3(SEQ / 128, TRIPLE / 128, BATCH), 256, 0, stream>>>(xT, wb, qT, kvb, ksum);
    k_context<<<dim3(HEADS, BATCH, 4), 128, 0, stream>>>(kvb, ctx);
    k_wfuse<<<(BATCH * DIM * DIM) / 256, 256, 0, stream>>>(wout, ctx, ksum, wfused);
    k_final<<<dim3(SEQ / 64, BATCH), 256, 0, stream>>>(qT, wfused, bout, g, out);
}
